// Decoder_52759378264445
// MI455X (gfx1250) — matrix operand order repaired and checked by reading, not compiled, not run on hardware
//
#include <hip/hip_runtime.h>
#include <hip/hip_bf16.h>

typedef __attribute__((ext_vector_type(16))) __bf16 v16bf;
typedef __attribute__((ext_vector_type(8)))  float  v8f;
typedef __attribute__((ext_vector_type(4)))  int    v4i;
typedef __attribute__((address_space(1))) v4i* gv4i_ptr;
typedef __attribute__((address_space(3))) v4i* lv4i_ptr;

#if defined(__has_builtin)
#  if __has_builtin(__builtin_amdgcn_global_load_async_to_lds_b128)
#    define HAVE_ASYNC_LDS 1
#  endif
#  if __has_builtin(__builtin_amdgcn_s_wait_asynccnt)
#    define HAVE_WAIT_ASYNC 1
#  endif
#endif

namespace {
constexpr int kT = 32, kB = 64, kS = 64, kH = 512, kE = 512;
constexpr int kG  = 2048;   // 4*H
constexpr int kK0 = 1536;   // (E+H) + H   (x | feed | h0)
constexpr int kK1 = 1024;   // H + H       (h0n | h1)
constexpr int NWG = 32, NTHR = 256;
constexpr int WPB = NTHR / 32;            // waves per block
constexpr int NWAVES = NWG * WPB;
constexpr int NT = NWG * NTHR;
constexpr int kLdsPad = 8;                // +16B/row -> conflict-free frag reads
constexpr int kLdsElems = 16 * (kK0 + kLdsPad);
// d_out float offsets: outputs | attns | hf | cf | feedf
constexpr int OB_ATT = kT * kB * kH;
constexpr int OB_HF  = OB_ATT + kT * kB * kS;
constexpr int OB_CF  = OB_HF + 2 * kB * kH;
constexpr int OB_FD  = OB_CF + 2 * kB * kH;
}

struct DecoderParams {
  const int*   tokens;
  const float *context, *embed;
  const float *bih0, *bhh0, *bih1, *bhh1;
  const float *Wih0, *Whh0, *Wih1, *Whh1, *Wa, *Wout;
  const float *h0_in, *c0_in, *feed_in;
  unsigned short *Wc0, *Wc1, *Wabf, *Woutbf;
  unsigned short *xcat0, *xcat1, *attn_in, *h1bf;
  float *gates0, *gates1, *qbuf;
  float *h0s, *h1s, *c0s, *c1s, *feed;
  float *dout;
  unsigned *bar;   // bar[0]=count, bar[1]=generation
};

__device__ __forceinline__ unsigned short f2bf(float f) {
  unsigned u = __builtin_bit_cast(unsigned, f);
  return (unsigned short)((u + 0x7FFFu + ((u >> 16) & 1u)) >> 16);
}
__device__ __forceinline__ float sigf(float x) { return 1.0f / (1.0f + __expf(-x)); }

// -------- device-wide sense barrier (persistent kernel) --------
__device__ __forceinline__ void grid_barrier(unsigned* bar) {
  __syncthreads();
  if (threadIdx.x == 0) {
    __threadfence();                                  // release
    unsigned g = __atomic_load_n(&bar[1], __ATOMIC_RELAXED);
    if (atomicAdd(&bar[0], 1u) == (unsigned)(NWG - 1)) {
      atomicExch(&bar[0], 0u);
      __atomic_store_n(&bar[1], g + 1u, __ATOMIC_RELEASE);
    } else {
      while (__atomic_load_n(&bar[1], __ATOMIC_ACQUIRE) == g)
        __builtin_amdgcn_s_sleep(1);
    }
    __threadfence();                                  // acquire
  }
  __syncthreads();
}

// B (32x16, K x N) = W^T with W stored [N,K] row-major: lane n=lane&15 holds
// column n, K runs of 8 at half*8 and 16+half*8 (ISA 7.12.2).
__device__ __forceinline__ v16bf load_b_frag(const unsigned short* W, int ldw,
                                             int nbase, int kbase) {
  const int lane = threadIdx.x & 31;
  const int n = lane & 15, half = lane >> 4;
  const unsigned short* p = W + (size_t)(nbase + n) * ldw + kbase + half * 8;
  union { uint4 q[2]; v16bf v; } u;
  u.q[0] = *(const uint4*)(p);
  u.q[1] = *(const uint4*)(p + 16);
  return u.v;
}

// C = A[64,K] * W[N,K]^T. Each wave owns one pair of adjacent 16-col tiles:
// one LDS A-fragment feeds two independent WMMA accumulate chains.
// All 8 waves of a block share one M-block, whose 16 x K A-rows are staged
// into LDS (async-to-LDS when available) once per call.
__device__ __forceinline__ void gemm_bf16_pair(
    const unsigned short* __restrict__ A, int lda,
    const unsigned short* __restrict__ W, int ldw,
    float* __restrict__ C, int ldc,
    int n_tiles, int K,
    unsigned short* smem, int gwave) {
  const int npairs = n_tiles >> 1;
  const int total  = 4 * npairs;          // 64 rows -> 4 M-blocks
  const int ldsK   = K + kLdsPad;
  const int wgFirst = (int)blockIdx.x * WPB;

  if (wgFirst < total) {                  // stage this WG's A rows
    const int bmS = wgFirst / npairs;
    const int rowChunks = K >> 3;         // 16B chunks per row
    for (int c = threadIdx.x; c < 16 * rowChunks; c += NTHR) {
      const int r = c / rowChunks, kc = c - r * rowChunks;
      const unsigned short* src = A + (size_t)(bmS * 16 + r) * lda + kc * 8;
      unsigned short* dst = smem + r * ldsK + kc * 8;
#if defined(HAVE_ASYNC_LDS)
      __builtin_amdgcn_global_load_async_to_lds_b128(
          (gv4i_ptr)src, (lv4i_ptr)dst, 0, 0);
#else
      *(uint4*)dst = *(const uint4*)src;
#endif
    }
#if defined(HAVE_ASYNC_LDS)
#  if defined(HAVE_WAIT_ASYNC)
    __builtin_amdgcn_s_wait_asynccnt(0);
#  else
    asm volatile("s_wait_asynccnt 0x0" ::: "memory");
#  endif
#endif
  }
  __syncthreads();

  if (gwave < total) {
    const int lane = threadIdx.x & 31;
    const int half = lane >> 4;
    const int bm  = gwave / npairs;
    const int n0  = (gwave % npairs) * 32;
    __builtin_prefetch(W + (size_t)n0 * ldw, 0, 1);
    v8f acc0 = {0.f, 0.f, 0.f, 0.f, 0.f, 0.f, 0.f, 0.f};
    v8f acc1 = {0.f, 0.f, 0.f, 0.f, 0.f, 0.f, 0.f, 0.f};
    const unsigned short* arow = smem + (lane & 15) * ldsK + half * 8;
    for (int kk = 0; kk < K; kk += 32) {
      union { uint4 q[2]; v16bf v; } ua;         // A frag from LDS
      ua.q[0] = *(const uint4*)(arow + kk);
      ua.q[1] = *(const uint4*)(arow + kk + 16);
      v16bf b0 = load_b_frag(W, ldw, n0, kk);
      v16bf b1 = load_b_frag(W, ldw, n0 + 16, kk);
      acc0 = __builtin_amdgcn_wmma_f32_16x16x32_bf16(false, ua.v, false, b0,
                                                     (short)0, acc0, false, false);
      acc1 = __builtin_amdgcn_wmma_f32_16x16x32_bf16(false, ua.v, false, b1,
                                                     (short)0, acc1, false, false);
    }
    const int n = n0 + (lane & 15);
#pragma unroll
    for (int j = 0; j < 8; ++j) {
      const int m = bm * 16 + j + half * 8;
      C[(size_t)m * ldc + n]      = acc0[j];
      C[(size_t)m * ldc + n + 16] = acc1[j];
    }
  }
}

// -------- prep: f32 -> bf16 weight repack + state init + barrier reset -------
__global__ void decoder_prep_kernel(DecoderParams p) {
  const int tid = blockIdx.x * blockDim.x + threadIdx.x;
  const int nt = gridDim.x * blockDim.x;
  if (tid == 0) { p.bar[0] = 0u; p.bar[1] = 0u; }
  for (int i = tid; i < kG * kK0; i += nt) {          // Wc0 = [Wih0 | Whh0]
    const int g = i / kK0, c = i - g * kK0;
    const float v = (c < 1024) ? p.Wih0[(size_t)g * 1024 + c]
                               : p.Whh0[(size_t)g * 512 + (c - 1024)];
    p.Wc0[i] = f2bf(v);
  }
  for (int i = tid; i < kG * kK1; i += nt) {          // Wc1 = [Wih1 | Whh1]
    const int g = i >> 10, c = i & 1023;
    const float v = (c < 512) ? p.Wih1[(size_t)g * 512 + c]
                              : p.Whh1[(size_t)g * 512 + (c - 512)];
    p.Wc1[i] = f2bf(v);
  }
  for (int i = tid; i < kH * kH; i += nt)      p.Wabf[i]   = f2bf(p.Wa[i]);
  for (int i = tid; i < kH * kK1; i += nt)     p.Woutbf[i] = f2bf(p.Wout[i]);
  for (int i = tid; i < kB * kH; i += nt) {
    p.h0s[i] = p.h0_in[i];           p.h1s[i] = p.h0_in[kB * kH + i];
    p.c0s[i] = p.c0_in[i];           p.c1s[i] = p.c0_in[kB * kH + i];
    p.feed[i] = p.feed_in[i];
  }
}

// -------- persistent decoder: 32 serial timesteps, grid-synced phases -------
__global__ __launch_bounds__(NTHR) void decoder_persistent_kernel(DecoderParams p) {
  __shared__ unsigned short smem[kLdsElems];
  const int tid = blockIdx.x * NTHR + threadIdx.x;
  const int gwave = tid >> 5;
  const int lane = threadIdx.x & 31;

  for (int t = 0; t < kT; ++t) {
    // P0: build xcat0 = bf16[emb | feed | h0]; xcat1 high half = bf16(h1)
    for (int i = tid; i < kB * kE; i += NT) {
      const int b = i >> 9, e = i & 511;
      const int tok = p.tokens[t * kB + b];
      p.xcat0[(size_t)b * kK0 + e] = f2bf(p.embed[(size_t)tok * kE + e]);
    }
    for (int i = tid; i < kB * kH; i += NT) {
      const int b = i >> 9, h = i & 511;
      p.xcat0[(size_t)b * kK0 + 512 + h]  = f2bf(p.feed[i]);
      p.xcat0[(size_t)b * kK0 + 1024 + h] = f2bf(p.h0s[i]);
      p.xcat1[(size_t)b * kK1 + 512 + h]  = f2bf(p.h1s[i]);
    }
    grid_barrier(p.bar);

    // P1: gates0 = xcat0 @ Wc0^T   [64,2048], K=1536
    gemm_bf16_pair(p.xcat0, kK0, p.Wc0, kK0, p.gates0, kG, kG / 16, kK0, smem, gwave);
    grid_barrier(p.bar);

    // P2: LSTM cell 0 elementwise -> h0s/c0s, xcat1 low half
    for (int i = tid; i < kB * kH; i += NT) {
      const int b = i >> 9, j = i & 511;
      const float* g = p.gates0 + (size_t)b * kG;
      const float gi = g[j]        + p.bih0[j]        + p.bhh0[j];
      const float gf = g[512 + j]  + p.bih0[512 + j]  + p.bhh0[512 + j];
      const float gg = g[1024 + j] + p.bih0[1024 + j] + p.bhh0[1024 + j];
      const float go = g[1536 + j] + p.bih0[1536 + j] + p.bhh0[1536 + j];
      const float c  = sigf(gf) * p.c0s[i] + sigf(gi) * tanhf(gg);
      const float h  = sigf(go) * tanhf(c);
      p.c0s[i] = c; p.h0s[i] = h;
      p.xcat1[(size_t)b * kK1 + j] = f2bf(h);
    }
    grid_barrier(p.bar);

    // P3: gates1 = xcat1 @ Wc1^T   [64,2048], K=1024
    gemm_bf16_pair(p.xcat1, kK1, p.Wc1, kK1, p.gates1, kG, kG / 16, kK1, smem, gwave);
    grid_barrier(p.bar);

    // P4: LSTM cell 1 elementwise -> h1s/c1s, h1bf, attn_in high half
    for (int i = tid; i < kB * kH; i += NT) {
      const int b = i >> 9, j = i & 511;
      const float* g = p.gates1 + (size_t)b * kG;
      const float gi = g[j]        + p.bih1[j]        + p.bhh1[j];
      const float gf = g[512 + j]  + p.bih1[512 + j]  + p.bhh1[512 + j];
      const float gg = g[1024 + j] + p.bih1[1024 + j] + p.bhh1[1024 + j];
      const float go = g[1536 + j] + p.bih1[1536 + j] + p.bhh1[1536 + j];
      const float c  = sigf(gf) * p.c1s[i] + sigf(gi) * tanhf(gg);
      const float h  = sigf(go) * tanhf(c);
      p.c1s[i] = c; p.h1s[i] = h;
      const unsigned short hb = f2bf(h);
      p.h1bf[i] = hb;
      p.attn_in[(size_t)b * kK1 + 512 + j] = hb;
    }
    grid_barrier(p.bar);

    // P5: q = h1n @ Wa^T   [64,512], K=512
    gemm_bf16_pair(p.h1bf, kH, p.Wabf, kH, p.qbuf, kH, kH / 16, kH, smem, gwave);
    grid_barrier(p.bar);

    // P6: attention scores + softmax + context vector (one wave per batch row)
    if (gwave < kB) {
      const int b = gwave;
      const float* qb = p.qbuf + (size_t)b * kH;
      float sc0 = 0.f, sc1 = 0.f;
      const float* c0p = p.context + ((size_t)lane * kB + b) * kH;
      const float* c1p = p.context + ((size_t)(lane + 32) * kB + b) * kH;
      for (int h = 0; h < kH; ++h) {
        const float qv = qb[h];
        sc0 += qv * c0p[h];
        sc1 += qv * c1p[h];
      }
      float m = fmaxf(sc0, sc1);
      for (int o = 16; o; o >>= 1) m = fmaxf(m, __shfl_xor(m, o, 32));
      const float e0 = __expf(sc0 - m), e1 = __expf(sc1 - m);
      float sum = e0 + e1;
      for (int o = 16; o; o >>= 1) sum += __shfl_xor(sum, o, 32);
      const float inv = 1.f / sum;
      const float a0 = e0 * inv, a1 = e1 * inv;
      p.dout[OB_ATT + (size_t)t * kB * kS + b * kS + lane]      = a0;
      p.dout[OB_ATT + (size_t)t * kB * kS + b * kS + lane + 32] = a1;
      float acc[16];
#pragma unroll
      for (int i = 0; i < 16; ++i) acc[i] = 0.f;
      for (int s = 0; s < kS; ++s) {
        const float av = __shfl(s < 32 ? a0 : a1, s & 31, 32);
        const float* crow = p.context + ((size_t)s * kB + b) * kH;
#pragma unroll
        for (int i = 0; i < 16; ++i) acc[i] += av * crow[lane + 32 * i];
      }
#pragma unroll
      for (int i = 0; i < 16; ++i)
        p.attn_in[(size_t)b * kK1 + lane + 32 * i] = f2bf(acc[i]);
    }
    grid_barrier(p.bar);

    // P7: aout = [cvec|h1n] @ Wout^T  [64,512], K=1024  (reuse qbuf)
    gemm_bf16_pair(p.attn_in, kK1, p.Woutbf, kK1, p.qbuf, kH, kH / 16, kK1, smem, gwave);
    grid_barrier(p.bar);

    // P8: attn_h = tanh(aout) -> outputs[t], feed
    for (int i = tid; i < kB * kH; i += NT) {
      const float v = tanhf(p.qbuf[i]);
      p.dout[(size_t)t * kB * kH + i] = v;
      p.feed[i] = v;
    }
    grid_barrier(p.bar);
  }

  // final states: hf, cf, feedf
  for (int i = tid; i < kB * kH; i += NT) {
    p.dout[OB_HF + i]           = p.h0s[i];
    p.dout[OB_HF + kB * kH + i] = p.h1s[i];
    p.dout[OB_CF + i]           = p.c0s[i];
    p.dout[OB_CF + kB * kH + i] = p.c1s[i];
    p.dout[OB_FD + i]           = p.feed[i];
  }
}

extern "C" void kernel_launch(void* const* d_in, const int* in_sizes, int n_in,
                              void* d_out, int out_size, void* d_ws, size_t ws_size,
                              hipStream_t stream) {
  (void)in_sizes; (void)n_in; (void)out_size; (void)ws_size;
  DecoderParams p;
  p.tokens  = (const int*)  d_in[0];
  // d_in[1] = src (unused by reference computation)
  p.context = (const float*)d_in[2];
  p.h0_in   = (const float*)d_in[3];
  p.c0_in   = (const float*)d_in[4];
  p.feed_in = (const float*)d_in[5];
  p.embed   = (const float*)d_in[6];
  p.Wih0    = (const float*)d_in[7];
  p.Whh0    = (const float*)d_in[8];
  p.bih0    = (const float*)d_in[9];
  p.bhh0    = (const float*)d_in[10];
  p.Wih1    = (const float*)d_in[11];
  p.Whh1    = (const float*)d_in[12];
  p.bih1    = (const float*)d_in[13];
  p.bhh1    = (const float*)d_in[14];
  p.Wa      = (const float*)d_in[15];
  p.Wout    = (const float*)d_in[16];
  p.dout    = (float*)d_out;

  char* base = (char*)d_ws; size_t off = 0;
  auto carve = [&](size_t bytes) -> char* {
    char* q = base + off; off += (bytes + 255) & ~(size_t)255; return q;
  };
  p.Wc0     = (unsigned short*)carve((size_t)kG * kK0 * 2);
  p.Wc1     = (unsigned short*)carve((size_t)kG * kK1 * 2);
  p.Wabf    = (unsigned short*)carve((size_t)kH * kH * 2);
  p.Woutbf  = (unsigned short*)carve((size_t)kH * kK1 * 2);
  p.xcat0   = (unsigned short*)carve((size_t)kB * kK0 * 2);
  p.xcat1   = (unsigned short*)carve((size_t)kB * kK1 * 2);
  p.attn_in = (unsigned short*)carve((size_t)kB * kK1 * 2);
  p.h1bf    = (unsigned short*)carve((size_t)kB * kH * 2);
  p.gates0  = (float*)carve((size_t)kB * kG * 4);
  p.gates1  = (float*)carve((size_t)kB * kG * 4);
  p.qbuf    = (float*)carve((size_t)kB * kH * 4);
  p.h0s     = (float*)carve((size_t)kB * kH * 4);
  p.h1s     = (float*)carve((size_t)kB * kH * 4);
  p.c0s     = (float*)carve((size_t)kB * kH * 4);
  p.c1s     = (float*)carve((size_t)kB * kH * 4);
  p.feed    = (float*)carve((size_t)kB * kH * 4);
  p.bar     = (unsigned*)carve(256);

  decoder_prep_kernel<<<256, NTHR, 0, stream>>>(p);
  decoder_persistent_kernel<<<NWG, NTHR, 0, stream>>>(p);
}
